// GATLayer_5643587027337
// MI455X (gfx1250) — compile-verified, matrix operand order checked by reading, not run on hardware
//
#include <hip/hip_runtime.h>

typedef __attribute__((ext_vector_type(16))) _Float16 v16h;
typedef __attribute__((ext_vector_type(8)))  _Float16 v8h;
typedef __attribute__((ext_vector_type(8)))  float    v8f;

#define NN   50000
#define EO   800000
#define EE   (EO + NN)      // edges + self loops = 850000
#define D    64
#define NEGS 0.2f
#define MT   5              // M tiles per block (3125 = 625 * 5)

// -------- helpers --------------------------------------------------------

__device__ __forceinline__ void edge_sd(const int* __restrict__ ei, int e,
                                        int& s, int& d) {
    if (e < EO) { s = ei[e]; d = ei[EO + e]; }
    else        { s = e - EO; d = s; }
}

__device__ __forceinline__ void atomicMaxF(float* addr, float val) {
    // monotone float max via signed/unsigned int atomics (init = -inf)
    if (val >= 0.0f) atomicMax((int*)addr, __float_as_int(val));
    else             atomicMin((unsigned int*)addr, __float_as_uint(val));
}

// -------- write ei output (edge_index with self loops, as float) ---------

__global__ void ei_kernel(const int* __restrict__ ei, float* __restrict__ out) {
    int tid = blockIdx.x * blockDim.x + threadIdx.x;
    if (tid >= 2 * EE) return;
    int row = tid / EE;
    int col = tid - row * EE;
    int v = (col < EO) ? ei[row * EO + col] : (col - EO);
    out[tid] = (float)v;
}

// -------- WMMA GEMM: H[NN x 64] = A[NN x KD] @ W[KD x 64] ----------------
// W staged once per block into LDS, transposed to f16 with padded stride so
// each B fragment is two aligned ds_load_b128's. One 16x16 tile per wave,
// 4 waves cover N=64, MT M-tiles per block. Full K unroll -> WMMA chain.

template<int KD, bool RELU>
__global__ __launch_bounds__(128)
void gemm_wmma(const float* __restrict__ A, const float* __restrict__ W,
               float* __restrict__ H) {
    constexpr int KP = KD + 8;              // half-stride pad: row = 16B*odd
    __shared__ __align__(16) _Float16 WT[64 * KP];

    const int t = threadIdx.x;
    // stage W transposed: WT[n][k] = (f16) W[k][n]; coalesced global reads
    for (int idx = t; idx < KD * 64; idx += 128) {
        int k = idx >> 6;
        int n = idx & 63;
        WT[n * KP + k] = (_Float16)W[idx];
    }
    __syncthreads();

    const int lane = t & 31;
    const int l16  = lane & 15;
    const int hi   = lane >> 4;             // K sub-base 0 / 8 (ISA layout)
    const int n0   = (t >> 5) * 16;         // wave -> N tile

    const _Float16* wt = &WT[(n0 + l16) * KP];

    for (int mt = 0; mt < MT; ++mt) {
        const int m0 = (blockIdx.x * MT + mt) * 16;
        v8f c = {};
#pragma unroll
        for (int k0 = 0; k0 < KD; k0 += 32) {
            // ---- A fragment: two contiguous 8-float runs, f32 -> f16 ----
            const float* arow = A + (size_t)(m0 + l16) * KD + k0 + hi * 8;
            float4 af0 = *(const float4*)(arow);
            float4 af1 = *(const float4*)(arow + 4);
            float4 af2 = *(const float4*)(arow + 16);
            float4 af3 = *(const float4*)(arow + 20);
            if (RELU) {
                af0.x = fmaxf(af0.x, 0.f); af0.y = fmaxf(af0.y, 0.f);
                af0.z = fmaxf(af0.z, 0.f); af0.w = fmaxf(af0.w, 0.f);
                af1.x = fmaxf(af1.x, 0.f); af1.y = fmaxf(af1.y, 0.f);
                af1.z = fmaxf(af1.z, 0.f); af1.w = fmaxf(af1.w, 0.f);
                af2.x = fmaxf(af2.x, 0.f); af2.y = fmaxf(af2.y, 0.f);
                af2.z = fmaxf(af2.z, 0.f); af2.w = fmaxf(af2.w, 0.f);
                af3.x = fmaxf(af3.x, 0.f); af3.y = fmaxf(af3.y, 0.f);
                af3.z = fmaxf(af3.z, 0.f); af3.w = fmaxf(af3.w, 0.f);
            }
            v16h a;
            a[0]  = (_Float16)af0.x; a[1]  = (_Float16)af0.y;
            a[2]  = (_Float16)af0.z; a[3]  = (_Float16)af0.w;
            a[4]  = (_Float16)af1.x; a[5]  = (_Float16)af1.y;
            a[6]  = (_Float16)af1.z; a[7]  = (_Float16)af1.w;
            a[8]  = (_Float16)af2.x; a[9]  = (_Float16)af2.y;
            a[10] = (_Float16)af2.z; a[11] = (_Float16)af2.w;
            a[12] = (_Float16)af3.x; a[13] = (_Float16)af3.y;
            a[14] = (_Float16)af3.z; a[15] = (_Float16)af3.w;

            // ---- B fragment: two aligned 16B LDS loads of WT column ----
            v8h b0 = *(const v8h*)(wt + k0 + hi * 8);
            v8h b1 = *(const v8h*)(wt + k0 + 16 + hi * 8);
            v16h b;
#pragma unroll
            for (int i = 0; i < 8; ++i) { b[i] = b0[i]; b[i + 8] = b1[i]; }

            c = __builtin_amdgcn_wmma_f32_16x16x32_f16(
                    /*neg_a=*/false, a, /*neg_b=*/false, b,
                    /*c_mod=*/(short)0, c, /*reuse_a=*/false, /*reuse_b=*/false);
        }
        // C/D layout: VGPR v -> row m0 + hi*8 + v, col n0 + l16
        float* hrow = H + (size_t)(m0 + hi * 8) * D + n0 + l16;
#pragma unroll
        for (int v = 0; v < 8; ++v)
            hrow[(size_t)v * D] = c[v];
    }
}

// -------- per-node attention logits: es = h.a_src, ed = h.a_dst ----------

__global__ __launch_bounds__(256)
void logits_kernel(const float* __restrict__ H, const float* __restrict__ as,
                   const float* __restrict__ ad, float* __restrict__ es,
                   float* __restrict__ ed) {
    int tid  = blockIdx.x * blockDim.x + threadIdx.x;
    int n    = tid >> 5;
    int lane = tid & 31;
    if (n >= NN) return;
    float h0 = H[(size_t)n * D + lane];
    float h1 = H[(size_t)n * D + lane + 32];
    float s = h0 * as[lane] + h1 * as[lane + 32];
    float d = h0 * ad[lane] + h1 * ad[lane + 32];
#pragma unroll
    for (int off = 16; off > 0; off >>= 1) {
        s += __shfl_xor(s, off, 32);
        d += __shfl_xor(d, off, 32);
    }
    if (lane == 0) { es[n] = s; ed[n] = d; }
}

// -------- init: agg = bias (broadcast), m = -inf, z = 0 ------------------

__global__ void init_kernel(float* __restrict__ agg, const float* __restrict__ bias,
                            float* __restrict__ m, float* __restrict__ z) {
    int tid = blockIdx.x * blockDim.x + threadIdx.x;
    if (tid < NN * D) agg[tid] = bias[tid & (D - 1)];
    if (tid < NN) { m[tid] = -__builtin_inff(); z[tid] = 0.0f; }
}

// -------- edge pass 1: leaky-relu logit + segment max --------------------

__global__ __launch_bounds__(256)
void edge_pass1(const int* __restrict__ ei, const float* __restrict__ es,
                const float* __restrict__ ed, float* __restrict__ ebuf,
                float* __restrict__ m) {
    int e = blockIdx.x * blockDim.x + threadIdx.x;
    if (e >= EE) return;
    int s, d; edge_sd(ei, e, s, d);
    float v = es[s] + ed[d];
    v = v > 0.0f ? v : NEGS * v;
    ebuf[e] = v;
    atomicMaxF(&m[d], v);
}

// -------- edge pass 2: exp(e - m[dst]) + segment sum ---------------------

__global__ __launch_bounds__(256)
void edge_pass2(const int* __restrict__ ei, float* __restrict__ ebuf,
                const float* __restrict__ m, float* __restrict__ z) {
    int e = blockIdx.x * blockDim.x + threadIdx.x;
    if (e >= EE) return;
    int s, d; edge_sd(ei, e, s, d);
    float ex = __expf(ebuf[e] - m[d]);
    ebuf[e] = ex;
    atomicAdd(&z[d], ex);
}

// -------- edge pass 3: alpha + weighted scatter-aggregation --------------
// one wave per edge; lane l handles channels l and l+32 (global f32 atomics)

__global__ __launch_bounds__(256)
void edge_pass3(const int* __restrict__ ei, const float* __restrict__ ebuf,
                const float* __restrict__ z, const float* __restrict__ H,
                float* __restrict__ agg, float* __restrict__ alpha_out) {
    int tid  = blockIdx.x * blockDim.x + threadIdx.x;
    int e    = tid >> 5;
    int lane = tid & 31;
    if (e >= EE) return;
    int s, d; edge_sd(ei, e, s, d);
    float alpha = ebuf[e] / z[d];
    if (alpha_out != nullptr && lane == 0) alpha_out[e] = alpha;
    float h0 = H[(size_t)s * D + lane];
    float h1 = H[(size_t)s * D + lane + 32];
    atomicAdd(&agg[(size_t)d * D + lane],      alpha * h0);
    atomicAdd(&agg[(size_t)d * D + lane + 32], alpha * h1);
}

// -------- host launcher --------------------------------------------------

extern "C" void kernel_launch(void* const* d_in, const int* in_sizes, int n_in,
                              void* d_out, int out_size, void* d_ws, size_t ws_size,
                              hipStream_t stream) {
    (void)in_sizes; (void)n_in; (void)out_size; (void)ws_size;

    const float* x    = (const float*)d_in[0];
    const int*   ei   = (const int*)  d_in[1];
    const float* W1   = (const float*)d_in[2];
    const float* as1  = (const float*)d_in[3];
    const float* ad1  = (const float*)d_in[4];
    const float* b1   = (const float*)d_in[5];
    const float* W2   = (const float*)d_in[6];
    const float* as2  = (const float*)d_in[7];
    const float* ad2  = (const float*)d_in[8];
    const float* b2   = (const float*)d_in[9];

    // output layout: [ei 2*EE | alpha1 EE | h2 NN*D]
    float* out       = (float*)d_out;
    float* ei_out    = out;
    float* alpha_out = out + 2 * (size_t)EE;
    float* h2_out    = alpha_out + (size_t)EE;

    // workspace layout (floats)
    float* ws   = (float*)d_ws;
    float* h1   = ws;                       // NN*D : x @ W1
    float* agg1 = h1   + (size_t)NN * D;    // NN*D : layer-1 aggregation
    float* h2g  = agg1 + (size_t)NN * D;    // NN*D : relu(agg1) @ W2
    float* es   = h2g  + (size_t)NN * D;    // NN
    float* ed   = es + NN;                  // NN
    float* m    = ed + NN;                  // NN : segment max
    float* z    = m  + NN;                  // NN : segment sum
    float* ebuf = z  + NN;                  // EE : edge scratch

    const int TB = 256;
    dim3 gEI((2 * EE + TB - 1) / TB);
    dim3 gGemm((NN / 16) / MT);             // 3125 tiles / 5 per block = 625
    dim3 gNode((NN * 32 + TB - 1) / TB);    // wave per node
    dim3 gInit((NN * D + TB - 1) / TB);
    dim3 gEdge((EE + TB - 1) / TB);
    dim3 gEdgeW(((size_t)EE * 32 + TB - 1) / TB); // wave per edge

    ei_kernel<<<gEI, TB, 0, stream>>>(ei, ei_out);

    // ---- layer 1 ----
    gemm_wmma<128, false><<<gGemm, 128, 0, stream>>>(x, W1, h1);
    logits_kernel<<<gNode, TB, 0, stream>>>(h1, as1, ad1, es, ed);
    init_kernel<<<gInit, TB, 0, stream>>>(agg1, b1, m, z);
    edge_pass1<<<gEdge, TB, 0, stream>>>(ei, es, ed, ebuf, m);
    edge_pass2<<<gEdge, TB, 0, stream>>>(ei, ebuf, m, z);
    edge_pass3<<<gEdgeW, TB, 0, stream>>>(ei, ebuf, z, h1, agg1, alpha_out);

    // ---- layer 2 (ReLU fused into GEMM A-load) ----
    gemm_wmma<64, true><<<gGemm, 128, 0, stream>>>(agg1, W2, h2g);
    logits_kernel<<<gNode, TB, 0, stream>>>(h2g, as2, ad2, es, ed);
    init_kernel<<<gInit, TB, 0, stream>>>(h2_out, b2, m, z);
    edge_pass1<<<gEdge, TB, 0, stream>>>(ei, es, ed, ebuf, m);
    edge_pass2<<<gEdge, TB, 0, stream>>>(ei, ebuf, m, z);
    edge_pass3<<<gEdgeW, TB, 0, stream>>>(ei, ebuf, z, h2g, h2_out, nullptr);
}
